// TransformerDecoderLayer_70153995813656
// MI455X (gfx1250) — matrix operand order repaired and checked by reading, not compiled, not run on hardware
//
#include <hip/hip_runtime.h>

// ---------------------------------------------------------------------------
// Transformer decoder layer (post-LN) for gfx1250 (MI455X), wave32 + WMMA.
// All matmuls run through v_wmma_f32_16x16x32_bf16 (bf16 in, f32 accumulate).
// GEMM A tiles are staged LDS-side by the Tensor Data Mover (TDM), double
// buffered, tracked with TENSORcnt. TDM use is UNCONDITIONAL this round so a
// successful compile proves tensor_load_to_lds is in the binary.
// ---------------------------------------------------------------------------

typedef __bf16 v16bf __attribute__((ext_vector_type(16)));
typedef float  v8f   __attribute__((ext_vector_type(8)));
typedef unsigned int u32x4 __attribute__((ext_vector_type(4)));
typedef int          i32x4 __attribute__((ext_vector_type(4)));
typedef int          i32x8 __attribute__((ext_vector_type(8)));

union FragAB { v16bf v; uint4 u[2]; };
union Pack4  { __bf16 b[4]; uint2 u; };

static constexpr int Bb   = 4;
static constexpr int Tt   = 1024;   // TQ == TK
static constexpr int Cc   = 1024;
static constexpr int Hh   = 16;
static constexpr int DFFf = 4096;
static constexpr int BT   = Bb * Tt;   // 4096 rows

__device__ __forceinline__ v8f vzero8() {
    v8f z;
#pragma unroll
    for (int i = 0; i < 8; ++i) z[i] = 0.0f;
    return z;
}

__device__ __forceinline__ v8f wmma_bf16(v16bf a, v16bf b, v8f c) {
    return __builtin_amdgcn_wmma_f32_16x16x32_bf16(
        /*neg_a=*/false, a, /*neg_b=*/false, b,
        /*c_mod=*/(short)0, c, /*reuse_a=*/false, /*reuse_b=*/false);
}

// TDM 2D tile load: tile_w x tile_h bf16 elements, row stride `stride_elems`
// in global, written to LDS with hardware padding of 4 DWORDs (16B) after
// every 16 DWORDs (64B) -> LDS row pitch of 80B == [..][40] bf16 layout.
__device__ __forceinline__ void tdm_load_tile_2d(unsigned lds_byte_addr,
                                                 const void* gptr,
                                                 unsigned tile_w,
                                                 unsigned tile_h,
                                                 unsigned stride_elems) {
    unsigned long long ga = (unsigned long long)(uintptr_t)gptr;
    u32x4 g0;
    g0[0] = 1u;                                    // count=1, user descriptor
    g0[1] = lds_byte_addr;                         // lds_addr
    g0[2] = (unsigned)(ga & 0xffffffffu);          // global_addr[31:0]
    g0[3] = (unsigned)((ga >> 32) & 0x01ffffffu)   // global_addr[56:32]
            | (2u << 30);                          // type = 2 ("image")
    i32x8 g1;
    g1[0] = (1 << 16)      // data_size = 2 bytes
          | (1 << 20)      // pad_enable
          | (3 << 22)      // pad_interval: 16 DWORDs (64B) of data
          | (3 << 25);     // pad_amount:   4 DWORDs (16B) of pad
    g1[1] = (int)((tile_w & 0xffffu) << 16);       // tensor_dim0[15:0]
    g1[2] = (int)(((tile_w >> 16) & 0xffffu)       // tensor_dim0[31:16]
          | ((tile_h & 0xffffu) << 16));           // tensor_dim1[15:0]
    g1[3] = (int)(((tile_h >> 16) & 0xffffu)       // tensor_dim1[31:16]
          | ((tile_w & 0xffffu) << 16));           // tile_dim0
    g1[4] = (int)(tile_h & 0xffffu);               // tile_dim1 (tile_dim2=0)
    g1[5] = (int)stride_elems;                     // tensor_dim0_stride[31:0]
    g1[6] = 0;                                     // stride[47:32] | dim1_stride
    g1[7] = 0;
    i32x4 z4; z4[0] = 0; z4[1] = 0; z4[2] = 0; z4[3] = 0;
#if defined(__clang_major__) && (__clang_major__ >= 23)
    i32x8 z8;
#pragma unroll
    for (int i = 0; i < 8; ++i) z8[i] = 0;
    __builtin_amdgcn_tensor_load_to_lds(g0, g1, z4, z4, z8, 0);
#else
    __builtin_amdgcn_tensor_load_to_lds(g0, g1, z4, z4, 0);
#endif
}

// ---------------------------------------------------------------------------
// f32 -> bf16 elementwise convert (vectorized x4)
// ---------------------------------------------------------------------------
__global__ void cvt_f32_bf16_k(const float* __restrict__ x,
                               __bf16* __restrict__ y, int n4) {
    int i = blockIdx.x * blockDim.x + threadIdx.x;
    if (i < n4) {
        float4 f = ((const float4*)x)[i];
        Pack4 p;
        p.b[0] = (__bf16)f.x; p.b[1] = (__bf16)f.y;
        p.b[2] = (__bf16)f.z; p.b[3] = (__bf16)f.w;
        ((uint2*)y)[i] = p.u;
    }
}

// ---------------------------------------------------------------------------
// Transpose-convert: w[K][N] f32 (row-major)  ->  wt[N][K] bf16 (row-major)
// ---------------------------------------------------------------------------
__global__ void transcvt_k(const float* __restrict__ w,
                           __bf16* __restrict__ wt, int K, int N) {
    __shared__ float t[32][33];
    const int kt = blockIdx.y * 32, nt = blockIdx.x * 32;
    const int tx = threadIdx.x, ty = threadIdx.y;   // 32 x 8
#pragma unroll
    for (int j = 0; j < 32; j += 8)
        t[ty + j][tx] = w[(size_t)(kt + ty + j) * N + nt + tx];
    __syncthreads();
#pragma unroll
    for (int j = 0; j < 32; j += 8)
        wt[(size_t)(nt + ty + j) * K + kt + tx] = (__bf16)t[tx][ty + j];
}

// ---------------------------------------------------------------------------
// GEMM:  out[M][N] = act( A[M][K](bf16) * Bt[N][K](bf16)^T + bias + resid )
// 256 threads = 8 waves; WG tile 128x128; wave tile 64x32 (4x2 WMMA accums).
// A tiles: TDM double-buffered into LDS (pitch 40 bf16 = 80B via HW padding).
// ACT: 0 = identity, 1 = exact GELU.
// ---------------------------------------------------------------------------
template <int ACT>
__global__ __launch_bounds__(256)
void gemm_bf16_k(const __bf16* __restrict__ A, const __bf16* __restrict__ Bt,
                 const float* __restrict__ bias, const float* __restrict__ resid,
                 float* __restrict__ outF, __bf16* __restrict__ outB,
                 int M, int N, int K) {
    __shared__ __align__(16) __bf16 a_lds[2][128][40];   // double buffered

    const int tid  = threadIdx.x;
    const int lane = tid & 31;
    const int wave = tid >> 5;
    const int hi   = (lane >> 4) & 1;
    const int l16  = lane & 15;
    const int wm   = wave & 1;     // 0..1 -> 64-row slab
    const int wn   = wave >> 1;    // 0..3 -> 32-col slab
    const int wgM  = blockIdx.y * 128;
    const int wgN  = blockIdx.x * 128;

    v8f acc[4][2];
#pragma unroll
    for (int mi = 0; mi < 4; ++mi)
#pragma unroll
        for (int ni = 0; ni < 2; ++ni) acc[mi][ni] = vzero8();

    // Prologue: kick off tile k0=0 into buffer 0.
    if (wave == 0) {
        tdm_load_tile_2d((unsigned)(uintptr_t)&a_lds[0][0][0],
                         A + (size_t)wgM * K, 32u, 128u, (unsigned)K);
    }

    int buf = 0;
    for (int k0 = 0; k0 < K; k0 += 32, buf ^= 1) {
        if (wave == 0) {
            if (k0 + 32 < K) {
                tdm_load_tile_2d((unsigned)(uintptr_t)&a_lds[buf ^ 1][0][0],
                                 A + (size_t)wgM * K + (k0 + 32),
                                 32u, 128u, (unsigned)K);
                __builtin_amdgcn_s_wait_tensorcnt((short)1);
            } else {
                __builtin_amdgcn_s_wait_tensorcnt((short)0);
            }
        }
        __syncthreads();   // TDM data visible to all waves

        if (k0 + 32 < K)
            __builtin_prefetch(
                Bt + (size_t)(wgN + tid / 2) * K + k0 + 32, 0, 0);

        FragAB af[4];
#pragma unroll
        for (int mi = 0; mi < 4; ++mi) {
            int row = wm * 64 + mi * 16 + l16;   // A-frag: M = lane%16
            int kb  = hi * 16;                   // K = 16*hi + s (contiguous)
            af[mi].u[0] = *(const uint4*)&a_lds[buf][row][kb];
            af[mi].u[1] = *(const uint4*)&a_lds[buf][row][kb + 8];
        }
        FragAB bfr[2];
#pragma unroll
        for (int ni = 0; ni < 2; ++ni) {
            int col = wgN + wn * 32 + ni * 16 + l16;  // B-frag: N = lane%16
            const uint4* p =
                (const uint4*)(Bt + (size_t)col * K + k0 + hi * 16);
            bfr[ni].u[0] = p[0];
            bfr[ni].u[1] = p[1];
        }
#pragma unroll
        for (int mi = 0; mi < 4; ++mi)
#pragma unroll
            for (int ni = 0; ni < 2; ++ni)
                acc[mi][ni] = wmma_bf16(af[mi].v, bfr[ni].v, acc[mi][ni]);

        __syncthreads();   // buffer safe to overwrite next iteration
    }

    // Epilogue: C/D layout -> M = r + 8*hi, N = lane%16
#pragma unroll
    for (int mi = 0; mi < 4; ++mi) {
#pragma unroll
        for (int ni = 0; ni < 2; ++ni) {
            int gn = wgN + wn * 32 + ni * 16 + l16;
            float bv = bias ? bias[gn] : 0.0f;
#pragma unroll
            for (int r = 0; r < 8; ++r) {
                int gm = wgM + wm * 64 + mi * 16 + r + hi * 8;
                float v = acc[mi][ni][r] + bv;
                size_t idx = (size_t)gm * N + gn;
                if (resid) v += resid[idx];
                if (ACT == 1)
                    v = 0.5f * v * (1.0f + erff(v * 0.70710678118654752f));
                if (outF) outF[idx] = v;
                if (outB) outB[idx] = (__bf16)v;
            }
        }
    }
}

// ---------------------------------------------------------------------------
// Flash attention (online softmax), bf16 operands, f32 accumulate.
// Grid: (T/64, H, B). 128 threads = 4 waves; wave w owns Q rows qt*64+w*16+..
// ---------------------------------------------------------------------------
__global__ __launch_bounds__(128)
void attn_k(const __bf16* __restrict__ Q, const __bf16* __restrict__ Kb,
            const __bf16* __restrict__ Vb, __bf16* __restrict__ O,
            int causal) {
    __shared__ __align__(16) __bf16 vt[64][40];       // V^T tile: [d][token]
    __shared__ __align__(16) __bf16 pl[4][16][40];    // per-wave P tile

    const int tid  = threadIdx.x;
    const int lane = tid & 31;
    const int wave = tid >> 5;
    const int hi   = (lane >> 4) & 1;
    const int l16  = lane & 15;
    const int qt   = blockIdx.x;
    const int h    = blockIdx.y;
    const int b    = blockIdx.z;

    // Preload Q A-fragments for both 32-wide halves of DK=64.
    const size_t rowQ = (size_t)(b * Tt + qt * 64 + wave * 16 + l16);
    FragAB qf[2];
#pragma unroll
    for (int ks = 0; ks < 2; ++ks) {
        const __bf16* p0 = Q + rowQ * Cc + h * 64 + ks * 32 + hi * 16;
        qf[ks].u[0] = *(const uint4*)p0;
        qf[ks].u[1] = *(const uint4*)(p0 + 8);
    }

    float mrow[8], lrow[8];
    v8f oacc[4];
#pragma unroll
    for (int r = 0; r < 8; ++r) { mrow[r] = -1e30f; lrow[r] = 0.0f; }
#pragma unroll
    for (int nt = 0; nt < 4; ++nt) oacc[nt] = vzero8();

    const int limit = causal ? (qt * 64 + 64) : Tt;

    for (int tk0 = 0; tk0 < limit; tk0 += 32) {
        // Stage V tile transposed into LDS: 32 tokens x 64 dims.
#pragma unroll
        for (int c0 = 0; c0 < 2; ++c0) {
            int c    = tid * 2 + c0;          // 256 chunks of 16B
            int vr   = c >> 3;                // token in tile
            int part = c & 7;                 // 8 dims per chunk
            const __bf16* src =
                Vb + (size_t)(b * Tt + tk0 + vr) * Cc + h * 64 + part * 8;
            uint4 d = *(const uint4*)src;
            const __bf16* dv = (const __bf16*)&d;
#pragma unroll
            for (int j = 0; j < 8; ++j) vt[part * 8 + j][vr] = dv[j];
        }
        __syncthreads();

        // Scores: S(16x32) = Q(16x64) * K_tile^T, two 16-col subtiles.
        v8f sacc[2];
#pragma unroll
        for (int nt = 0; nt < 2; ++nt) {
            sacc[nt] = vzero8();
#pragma unroll
            for (int ks = 0; ks < 2; ++ks) {
                FragAB kf;
                const __bf16* p0 =
                    Kb + (size_t)(b * Tt + tk0 + nt * 16 + l16) * Cc +
                    h * 64 + ks * 32 + hi * 16;
                kf.u[0] = ((const uint4*)p0)[0];
                kf.u[1] = ((const uint4*)p0)[1];
                sacc[nt] = wmma_bf16(qf[ks].v, kf.v, sacc[nt]);
            }
        }

        // Online softmax (rows striped: VGPR r -> M = r + 8*hi, N = lane%16).
#pragma unroll
        for (int r = 0; r < 8; ++r) {
            int tq = qt * 64 + wave * 16 + r + hi * 8;
            float s0 = sacc[0][r] * 0.125f;               // 1/sqrt(64)
            float s1 = sacc[1][r] * 0.125f;
            if (causal) {
                if (tk0 + l16 > tq)      s0 = -1e30f;
                if (tk0 + 16 + l16 > tq) s1 = -1e30f;
            }
            float tm = fmaxf(s0, s1);
#pragma unroll
            for (int mks = 1; mks < 16; mks <<= 1)        // within 16-lane half
                tm = fmaxf(tm, __shfl_xor(tm, mks));
            float mnew = fmaxf(mrow[r], tm);
            float corr = __expf(mrow[r] - mnew);
            mrow[r] = mnew;
            float p0 = __expf(s0 - mnew);
            float p1 = __expf(s1 - mnew);
            float ps = p0 + p1;
#pragma unroll
            for (int mks = 1; mks < 16; mks <<= 1)
                ps += __shfl_xor(ps, mks);
            lrow[r] = lrow[r] * corr + ps;
#pragma unroll
            for (int nt = 0; nt < 4; ++nt) oacc[nt][r] *= corr;
            // C-layout -> LDS (row-major P tile) for A-fragment re-read.
            pl[wave][r + hi * 8][l16]      = (__bf16)p0;
            pl[wave][r + hi * 8][16 + l16] = (__bf16)p1;
        }

        // P A-fragment (wave-private LDS round trip; DScnt keeps order).
        FragAB pf;
        {
            int kb = hi * 16;
            pf.u[0] = *(const uint4*)&pl[wave][l16][kb];
            pf.u[1] = *(const uint4*)&pl[wave][l16][kb + 8];
        }
        // O(16x64) += P(16x32) * V(32x64): 4 WMMAs over d-subtiles.
#pragma unroll
        for (int nt = 0; nt < 4; ++nt) {
            FragAB vf;
            vf.u[0] = *(const uint4*)&vt[nt * 16 + l16][hi * 16];
            vf.u[1] = *(const uint4*)&vt[nt * 16 + l16][hi * 16 + 8];
            oacc[nt] = wmma_bf16(pf.v, vf.v, oacc[nt]);
        }
        __syncthreads();   // protect vt before next tile overwrites it
    }

    // Normalize + store
#pragma unroll
    for (int r = 0; r < 8; ++r) {
        float inv = 1.0f / lrow[r];
        int grow = b * Tt + qt * 64 + wave * 16 + r + hi * 8;
#pragma unroll
        for (int nt = 0; nt < 4; ++nt) {
            int gcol = h * 64 + nt * 16 + l16;
            O[(size_t)grow * Cc + gcol] = (__bf16)(oacc[nt][r] * inv);
        }
    }
}

// ---------------------------------------------------------------------------
// LayerNorm over C=1024: y = (x - mu) * rsqrt(var + eps) * g + b
// ---------------------------------------------------------------------------
__global__ __launch_bounds__(256)
void ln_k(const float* __restrict__ x, const float* __restrict__ g,
          const float* __restrict__ bta, float* __restrict__ yf,
          __bf16* __restrict__ yb) {
    __shared__ float red[256];
    const int row = blockIdx.x;
    const float* xr = x + (size_t)row * Cc;
    float v[4], s = 0.0f, s2 = 0.0f;
#pragma unroll
    for (int j = 0; j < 4; ++j) {
        v[j] = xr[threadIdx.x + j * 256];
        s += v[j];
        s2 += v[j] * v[j];
    }
    red[threadIdx.x] = s;
    __syncthreads();
    for (int off = 128; off > 0; off >>= 1) {
        if (threadIdx.x < off) red[threadIdx.x] += red[threadIdx.x + off];
        __syncthreads();
    }
    float mean = red[0] * (1.0f / Cc);
    __syncthreads();
    red[threadIdx.x] = s2;
    __syncthreads();
    for (int off = 128; off > 0; off >>= 1) {
        if (threadIdx.x < off) red[threadIdx.x] += red[threadIdx.x + off];
        __syncthreads();
    }
    float var = red[0] * (1.0f / Cc) - mean * mean;
    float rs = rsqrtf(var + 1e-12f);
#pragma unroll
    for (int j = 0; j < 4; ++j) {
        int c = threadIdx.x + j * 256;
        float o = (v[j] - mean) * rs * g[c] + bta[c];
        if (yf) yf[(size_t)row * Cc + c] = o;
        if (yb) yb[(size_t)row * Cc + c] = (__bf16)o;
    }
}

// ---------------------------------------------------------------------------
// Orchestration
// ---------------------------------------------------------------------------
static void launch_cvt(const float* x, __bf16* y, size_t n, hipStream_t s) {
    int n4 = (int)(n / 4);
    cvt_f32_bf16_k<<<(n4 + 255) / 256, 256, 0, s>>>(x, y, n4);
}
static void launch_trans(const float* w, __bf16* wt, int K, int N,
                         hipStream_t s) {
    transcvt_k<<<dim3(N / 32, K / 32), dim3(32, 8), 0, s>>>(w, wt, K, N);
}
template <int ACT>
static void launch_gemm(const __bf16* A, const __bf16* Bt, const float* bias,
                        const float* resid, float* outF, __bf16* outB,
                        int M, int N, int K, hipStream_t s) {
    gemm_bf16_k<ACT><<<dim3(N / 128, M / 128), 256, 0, s>>>(
        A, Bt, bias, resid, outF, outB, M, N, K);
}

extern "C" void kernel_launch(void* const* d_in, const int* in_sizes, int n_in,
                              void* d_out, int out_size, void* d_ws,
                              size_t ws_size, hipStream_t stream) {
    (void)in_sizes; (void)n_in; (void)out_size; (void)ws_size;
    const float* src    = (const float*)d_in[0];
    const float* dst    = (const float*)d_in[1];
    const float* sa_wq  = (const float*)d_in[4];
    const float* sa_wk  = (const float*)d_in[5];
    const float* sa_wv  = (const float*)d_in[6];
    const float* sa_wo  = (const float*)d_in[7];
    const float* sa_bq  = (const float*)d_in[8];
    const float* sa_bk  = (const float*)d_in[9];
    const float* sa_bv  = (const float*)d_in[10];
    const float* sa_bo  = (const float*)d_in[11];
    const float* ca_wq  = (const float*)d_in[12];
    const float* ca_wk  = (const float*)d_in[13];
    const float* ca_wv  = (const float*)d_in[14];
    const float* ca_wo  = (const float*)d_in[15];
    const float* ca_bq  = (const float*)d_in[16];
    const float* ca_bk  = (const float*)d_in[17];
    const float* ca_bv  = (const float*)d_in[18];
    const float* ca_bo  = (const float*)d_in[19];
    const float* ffn_w1 = (const float*)d_in[20];
    const float* ffn_b1 = (const float*)d_in[21];
    const float* ffn_w2 = (const float*)d_in[22];
    const float* ffn_b2 = (const float*)d_in[23];
    const float* ln1_g  = (const float*)d_in[24];
    const float* ln1_b  = (const float*)d_in[25];
    const float* ln2_g  = (const float*)d_in[26];
    const float* ln2_b  = (const float*)d_in[27];
    const float* ln3_g  = (const float*)d_in[28];
    const float* ln3_b  = (const float*)d_in[29];
    float* out = (float*)d_out;

    char* wp = (char*)d_ws;
    auto alloc = [&](size_t bytes) -> char* {
        char* r = wp;
        wp += (bytes + 255) & ~size_t(255);
        return r;
    };
    const size_t NA = (size_t)BT * Cc;          // 4M elements
    __bf16* dst_bf  = (__bf16*)alloc(NA * 2);
    __bf16* src_bf  = (__bf16*)alloc(NA * 2);
    __bf16* wt      = (__bf16*)alloc((size_t)Cc * DFFf * 2);  // max weight
    __bf16* q_bf    = (__bf16*)alloc(NA * 2);
    __bf16* k_bf    = (__bf16*)alloc(NA * 2);
    __bf16* v_bf    = (__bf16*)alloc(NA * 2);
    __bf16* attn_bf = (__bf16*)alloc(NA * 2);
    float*  h_f     = (float*)alloc(NA * 4);
    float*  y_f     = (float*)alloc(NA * 4);
    __bf16* y_bf    = (__bf16*)alloc(NA * 2);
    __bf16* g_bf    = (__bf16*)alloc((size_t)BT * DFFf * 2);

    dim3 agrid(Tt / 64, Hh, Bb);

    // ---- Self-attention (causal) ----
    launch_cvt(dst, dst_bf, NA, stream);
    launch_cvt(src, src_bf, NA, stream);
    launch_trans(sa_wq, wt, Cc, Cc, stream);
    launch_gemm<0>(dst_bf, wt, sa_bq, nullptr, nullptr, q_bf, BT, Cc, Cc, stream);
    launch_trans(sa_wk, wt, Cc, Cc, stream);
    launch_gemm<0>(dst_bf, wt, sa_bk, nullptr, nullptr, k_bf, BT, Cc, Cc, stream);
    launch_trans(sa_wv, wt, Cc, Cc, stream);
    launch_gemm<0>(dst_bf, wt, sa_bv, nullptr, nullptr, v_bf, BT, Cc, Cc, stream);
    attn_k<<<agrid, 128, 0, stream>>>(q_bf, k_bf, v_bf, attn_bf, 1);
    launch_trans(sa_wo, wt, Cc, Cc, stream);
    launch_gemm<0>(attn_bf, wt, sa_bo, dst, h_f, nullptr, BT, Cc, Cc, stream);
    ln_k<<<BT, 256, 0, stream>>>(h_f, ln1_g, ln1_b, y_f, y_bf);

    // ---- Cross-attention (no mask) ----
    launch_trans(ca_wq, wt, Cc, Cc, stream);
    launch_gemm<0>(y_bf, wt, ca_bq, nullptr, nullptr, q_bf, BT, Cc, Cc, stream);
    launch_trans(ca_wk, wt, Cc, Cc, stream);
    launch_gemm<0>(src_bf, wt, ca_bk, nullptr, nullptr, k_bf, BT, Cc, Cc, stream);
    launch_trans(ca_wv, wt, Cc, Cc, stream);
    launch_gemm<0>(src_bf, wt, ca_bv, nullptr, nullptr, v_bf, BT, Cc, Cc, stream);
    attn_k<<<agrid, 128, 0, stream>>>(q_bf, k_bf, v_bf, attn_bf, 0);
    launch_trans(ca_wo, wt, Cc, Cc, stream);
    launch_gemm<0>(attn_bf, wt, ca_bo, y_f, h_f, nullptr, BT, Cc, Cc, stream);
    ln_k<<<BT, 256, 0, stream>>>(h_f, ln2_g, ln2_b, y_f, y_bf);

    // ---- FFN ----
    launch_trans(ffn_w1, wt, Cc, DFFf, stream);
    launch_gemm<1>(y_bf, wt, ffn_b1, nullptr, nullptr, g_bf, BT, DFFf, Cc, stream);
    launch_trans(ffn_w2, wt, DFFf, Cc, stream);
    launch_gemm<0>(g_bf, wt, ffn_b2, y_f, h_f, nullptr, BT, Cc, DFFf, stream);
    ln_k<<<BT, 256, 0, stream>>>(h_f, ln3_g, ln3_b, out, nullptr);
}
